// FSQVAE_47029891891261
// MI455X (gfx1250) — matrix operand order repaired and checked by reading, not compiled, not run on hardware
//
#include <hip/hip_runtime.h>
#include <hip/hip_bf16.h>

// ---------------------------------------------------------------------------
// FSQ-VAE forward for gfx1250 (MI455X).
// conv1d layers = implicit GEMM through v_wmma_f32_16x16x32_bf16 (bf16 A/B,
// fp32 accumulate). K order = ci*ks + kk == natural [Cout][Cin*ks] layout.
// Weights are repacked fp32->bf16 once per launch (if ws allows), so the hot
// loop A path is pure b128 load -> b128 LDS store. ReLU / residual /
// dilation / stride-2 / x2-upsample / output transpose fused into gather +
// epilogue.
// ---------------------------------------------------------------------------

typedef __attribute__((ext_vector_type(16))) __bf16 v16bf;
typedef __attribute__((ext_vector_type(8)))  __bf16 v8bf;
typedef __attribute__((ext_vector_type(8)))  float  v8f;

struct BfPair { v8bf lo, hi; };   // 32 bytes == sizeof(v16bf)

__device__ __forceinline__ unsigned short f2bf(float f) {
  unsigned u = __builtin_bit_cast(unsigned, f);
  u += 0x7FFFu + ((u >> 16) & 1u);            // round-to-nearest-even
  return (unsigned short)(u >> 16);
}
__device__ __forceinline__ float bf2f(unsigned short h) {
  unsigned u = ((unsigned)h) << 16;
  return __builtin_bit_cast(float, u);
}

enum { F_PRERELU = 1, F_POSTRELU = 2, F_UP2 = 4, F_TROUT = 8, F_WBF16 = 16 };

struct ConvArgs {
  const unsigned short* X;    // bf16 activations [B][Cin][Lin]
  const float*          W;    // fp32 weights [Cout][Cin*ks] (fallback path)
  const unsigned short* Wb;   // bf16 weights (fast path) or null
  const float*          Bias; // fp32 [Cout]
  const unsigned short* Res;  // bf16 residual [B][Cout][Lout] or null
  unsigned short*       Yb;   // bf16 out or null
  float*                Yf;   // fp32 out or null
  int Cin, Cout, Lin, Lout, ks, stride, pad, dil, flags;
};

// 4 waves; workgroup computes a 64(M=Cout) x 64(N=time) tile, K-step 32.
__launch_bounds__(128)
__global__ void conv1d_bf16_wmma(ConvArgs a) {
  __shared__ unsigned short As[64][40];   // [m][k]  (pad 40 -> conflict-free b128)
  __shared__ unsigned short Bs[64][40];   // [n][k]  (transposed gather tile)

  const int tid  = threadIdx.x;
  const int lane = tid & 31;
  const int wave = tid >> 5;
  const int b      = blockIdx.z;
  const int coBase = blockIdx.y * 64;
  const int nBase  = blockIdx.x * 64;

  const int up2    = (a.flags & F_UP2) ? 1 : 0;
  const int LinEff = a.Lin << up2;
  const int Ktot   = a.Cin * a.ks;

  v8f acc[4] = {};                               // 4 x (16x16 f32) per wave

  const int row   = tid >> 1;                    // 0..63 (both tiles)
  const int kcol0 = (tid & 1) * 16;              // 0 or 16
  const int t_out = nBase + row;
  const int xbase = b * a.Cin * a.Lin;
  const int wofs  = (coBase + row) * Ktot + kcol0;
  const int tb = t_out * a.stride - a.pad;       // gather base (pre-dilation)
  const bool nOK = (t_out < a.Lout);

  for (int kb = 0; kb < Ktot; kb += 32) {
    // ---- A tile: bf16 fast path (2 x b128), fp32 fallback (4 x b128+cvt) --
    if (a.flags & F_WBF16) {
      const uint4* wp = (const uint4*)(a.Wb + wofs + kb);
      *(uint4*)&As[row][kcol0]     = wp[0];
      *(uint4*)&As[row][kcol0 + 8] = wp[1];
    } else {
      const float4* wp = (const float4*)(a.W + wofs + kb);
#pragma unroll
      for (int q = 0; q < 4; ++q) {
        float4 f = wp[q];
        As[row][kcol0 + q * 4 + 0] = f2bf(f.x);
        As[row][kcol0 + q * 4 + 1] = f2bf(f.y);
        As[row][kcol0 + q * 4 + 2] = f2bf(f.z);
        As[row][kcol0 + q * 4 + 3] = f2bf(f.w);
      }
    }
    // ---- B tile: gathered input (pad/dilation/stride/upsample/preReLU) ----
    {
      int kr0 = kb + kcol0;
      int ci, kk;
      if (a.ks == 1)      { ci = kr0;      kk = 0; }
      else if (a.ks == 4) { ci = kr0 >> 2; kk = kr0 & 3; }
      else                { ci = (int)((unsigned)kr0 / 3u); kk = kr0 - ci * 3; }
#pragma unroll
      for (int j = 0; j < 16; ++j) {
        int ts = tb + kk * a.dil;
        unsigned short v = 0;
        if (nOK && ts >= 0 && ts < LinEff) {
          v = a.X[xbase + ci * a.Lin + (ts >> up2)];
          if ((a.flags & F_PRERELU) && (v & 0x8000u)) v = 0;   // bf16 relu
        }
        Bs[row][kcol0 + j] = v;
        if (++kk == a.ks) { kk = 0; ++ci; }
      }
    }
    __syncthreads();

    // ---- fragments per ISA 7.12.2 and 4 WMMAs per wave ----
    const int m    = lane & 15;
    const int half = lane >> 4;
    BfPair pa{ *(const v8bf*)&As[wave * 16 + m][half * 8],
               *(const v8bf*)&As[wave * 16 + m][16 + half * 8] };
    v16bf afrag = __builtin_bit_cast(v16bf, pa);
#pragma unroll
    for (int nf = 0; nf < 4; ++nf) {
      const unsigned short* bp = &Bs[nf * 16 + m][half * 8];
      BfPair pb{ *(const v8bf*)bp, *(const v8bf*)(bp + 16) };
      v16bf bfrag = __builtin_bit_cast(v16bf, pb);
      acc[nf] = __builtin_amdgcn_wmma_f32_16x16x32_bf16(
          false, afrag, false, bfrag, (short)0, acc[nf], false, false);
    }
    __syncthreads();
  }

  // ---- epilogue: bias + relu + residual, bf16/f32/transposed stores ----
  const int m8 = (lane >> 4) * 8;
  const int nc = lane & 15;
  const int co0 = coBase + wave * 16 + m8;       // 8 consecutive rows per lane
  float4 bia0 = *(const float4*)&a.Bias[co0];
  float4 bia1 = *(const float4*)&a.Bias[co0 + 4];
  float bias8[8] = { bia0.x, bia0.y, bia0.z, bia0.w, bia1.x, bia1.y, bia1.z, bia1.w };
#pragma unroll
  for (int nf = 0; nf < 4; ++nf) {
    int t = nBase + nf * 16 + nc;
    if (t >= a.Lout) continue;
#pragma unroll
    for (int r = 0; r < 8; ++r) {
      int co = co0 + r;
      float v = acc[nf][r] + bias8[r];
      if (a.flags & F_POSTRELU) v = fmaxf(v, 0.0f);
      int oidx = (b * a.Cout + co) * a.Lout + t;
      if (a.Res) v += bf2f(a.Res[oidx]);
      if (a.Yb) a.Yb[oidx] = f2bf(v);
      if (a.Yf) {
        if (a.flags & F_TROUT) a.Yf[(b * a.Lout + t) * a.Cout + co] = v;
        else                   a.Yf[oidx] = v;
      }
    }
  }
}

// ---- weight repack: flat fp32 -> bf16 (layout already GEMM-ready) ---------
__launch_bounds__(256)
__global__ void w_to_bf16(const float* W, unsigned short* Wb, int n) {
  int i = blockIdx.x * 256 + threadIdx.x;
  if (i < n) Wb[i] = f2bf(W[i]);
}

// ---- features (B,1024,64) f32 -> bf16 (B,64,1024) -------------------------
__launch_bounds__(256)
__global__ void feat_to_bf16(const float* F, unsigned short* X) {
  int i = blockIdx.x * 256 + threadIdx.x;
  if (i >= 4 * 64 * 1024) return;
  int t = i & 1023, c = (i >> 10) & 63, b = i >> 16;
  X[i] = f2bf(F[(b * 1024 + t) * 64 + c]);
}

// ---- FSQ 2-stage quantize + rotation trick (norms over time axis, 128) ----
__launch_bounds__(128)
__global__ void fsq_rot(const float* xenc, const float* levels,
                        unsigned short* xq, unsigned* used) {
  __shared__ float    red[128];
  __shared__ unsigned um;
  const int t  = threadIdx.x;
  const int bc = blockIdx.x;            // (b*512 + c), 2048 blocks
  float l[4] = { levels[0], levels[1], levels[2], levels[3] };
  float x = xenc[bc * 128 + t];

  int i1 = 0; float bd = fabsf(x - l[0]);
  for (int j = 1; j < 4; ++j) { float d = fabsf(x - l[j]); if (d < bd) { bd = d; i1 = j; } }
  float q1 = l[i1];
  float r  = x - q1;
  int i2 = 0; float bd2 = fabsf(r - l[0]);
  for (int j = 1; j < 4; ++j) { float d = fabsf(r - l[j]); if (d < bd2) { bd2 = d; i2 = j; } }
  float xqv = q1 + l[i2];

  if (t == 0) um = 0u;
  __syncthreads();
  atomicOr(&um, 1u << i1);

  auto bsum = [&](float v) -> float {
    red[t] = v; __syncthreads();
    for (int s = 64; s > 0; s >>= 1) { if (t < s) red[t] += red[t + s]; __syncthreads(); }
    float s0 = red[0]; __syncthreads();
    return s0;
  };

  float nx = sqrtf(bsum(x * x));
  float nq = sqrtf(bsum(xqv * xqv));
  float e  = x   / fmaxf(nx, 1e-12f);
  float qn = xqv / fmaxf(nq, 1e-12f);
  float s  = e + qn;
  float nr = sqrtf(bsum(s * s));
  float rr = s / fmaxf(nr, 1e-12f);
  float drx = bsum(rr * x);
  float dex = bsum(e * x);
  float rot = x - 2.0f * rr * drx + 2.0f * qn * dex;
  xq[bc * 128 + t] = f2bf((nq / nx) * rot);
  if (t == 0) atomicOr(used, um);
}

__global__ void init_mask(unsigned* m) { if (threadIdx.x == 0) *m = 0u; }

__global__ void finalize_k(float* out, const unsigned* used, int base) {
  if (threadIdx.x == 0 && blockIdx.x == 0) {
    out[base]     = 0.0f;                                   // commit
    out[base + 1] = 25.0f * (float)__popc(*used & 0xFu);    // 100*mean(used[4])
  }
}

// ---------------------------------------------------------------------------
// Host side
// ---------------------------------------------------------------------------
static void launch_conv(hipStream_t st,
                        const unsigned short* X, const float* W,
                        const unsigned short* Wb, const float* B,
                        const unsigned short* Res, unsigned short* Yb, float* Yf,
                        int Cin, int Cout, int Lin, int Lout,
                        int ks, int stride, int pad, int dil, int flags) {
  if (Wb) flags |= F_WBF16;
  ConvArgs a{ X, W, Wb, B, Res, Yb, Yf, Cin, Cout, Lin, Lout,
              ks, stride, pad, dil, flags };
  dim3 g((unsigned)((Lout + 63) / 64), (unsigned)(Cout / 64), 4u);
  conv1d_bf16_wmma<<<g, dim3(128), 0, st>>>(a);
}

extern "C" void kernel_launch(void* const* d_in, const int* in_sizes, int n_in,
                              void* d_out, int out_size, void* d_ws, size_t ws_size,
                              hipStream_t stream) {
  (void)n_in;
  // --- input index maps: insertion-order vs jax sorted-key flattening ------
  const bool ins = (in_sizes[0] == 4 * 1024 * 64);
  int iFeat, iLev, eInW, eInB, eOutW, eOutB, dInW, dInB, dMidW, dMidB, dOutW, dOutB;
  int eDownW[3], eDownB[3], eC1W[3][3], eC1B[3][3], eC2W[3][3], eC2B[3][3];
  int dUpW[3],  dUpB[3],   dC1W[3][3], dC1B[3][3], dC2W[3][3], dC2B[3][3];
  if (ins) {
    iFeat = 0; eInW = 1; eInB = 2;
    for (int i = 0; i < 3; ++i) {
      int base = 3 + i * 14;
      eDownW[i] = base; eDownB[i] = base + 1;
      for (int j = 0; j < 3; ++j) {
        int rb = base + 2 + j * 4;
        eC1W[i][j] = rb; eC1B[i][j] = rb + 1; eC2W[i][j] = rb + 2; eC2B[i][j] = rb + 3;
      }
    }
    eOutW = 45; eOutB = 46; dInW = 47; dInB = 48;
    for (int i = 0; i < 3; ++i) {
      int base = 49 + i * 14;
      for (int j = 0; j < 3; ++j) {
        int rb = base + j * 4;
        dC1W[i][j] = rb; dC1B[i][j] = rb + 1; dC2W[i][j] = rb + 2; dC2B[i][j] = rb + 3;
      }
      dUpW[i] = base + 12; dUpB[i] = base + 13;
    }
    dMidW = 91; dMidB = 92; dOutW = 93; dOutB = 94; iLev = 95;
  } else {           // sorted keys: dec_params, enc_params, features, fsq_levels; {'b','w'}
    for (int i = 0; i < 3; ++i) {
      int base = i * 14;
      for (int j = 0; j < 3; ++j) {
        int rb = base + j * 4;
        dC1B[i][j] = rb; dC1W[i][j] = rb + 1; dC2B[i][j] = rb + 2; dC2W[i][j] = rb + 3;
      }
      dUpB[i] = base + 12; dUpW[i] = base + 13;
    }
    dInB = 42; dInW = 43; dMidB = 44; dMidW = 45; dOutB = 46; dOutW = 47;
    for (int i = 0; i < 3; ++i) {
      int base = 48 + i * 14;
      eDownB[i] = base; eDownW[i] = base + 1;
      for (int j = 0; j < 3; ++j) {
        int rb = base + 2 + j * 4;
        eC1B[i][j] = rb; eC1W[i][j] = rb + 1; eC2B[i][j] = rb + 2; eC2W[i][j] = rb + 3;
      }
    }
    eInB = 90; eInW = 91; eOutB = 92; eOutW = 93; iFeat = 94; iLev = 95;
  }
  auto Wp = [&](int i) { return (const float*)d_in[i]; };

  // --- workspace layout ----------------------------------------------------
  char* w = (char*)d_ws;
  unsigned short* act0 = (unsigned short*)(w);                        // 4 MB
  unsigned short* act1 = (unsigned short*)(w + (4u << 20));           // 4 MB
  unsigned short* act2 = (unsigned short*)(w + (8u << 20));           // 4 MB
  float*          xenc = (float*)(w + (12u << 20));                   // 1 MB
  unsigned*       used = (unsigned*)(w + (13u << 20) + (1u << 19));

  // bf16 weight pool (52 MB) starting at 14 MB, if workspace permits.
  const size_t W_TOTAL = 26935296;                 // total weight elements
  size_t wOff = (size_t)14u << 20;
  const bool useWb = (ws_size >= wOff + W_TOTAL * 2);
  auto WB = [&](int idx) -> const unsigned short* { // repack one weight tensor
    if (!useWb) return nullptr;
    int n = in_sizes[idx];
    unsigned short* dst = (unsigned short*)(w + wOff);
    wOff += (size_t)n * 2;
    w_to_bf16<<<(n + 255) / 256, 256, 0, stream>>>(Wp(idx), dst, n);
    return dst;
  };

  init_mask<<<1, 32, 0, stream>>>(used);
  feat_to_bf16<<<(4 * 64 * 1024 + 255) / 256, 256, 0, stream>>>((const float*)d_in[iFeat], act0);

  const int ENC_DIL[3] = { 1, 3, 9 };
  const int DEC_DIL[3] = { 9, 3, 1 };
  unsigned short *cur, *t1, *t2;
  auto step = [&]() { unsigned short* o = cur; cur = t1; t1 = t2; t2 = o; };

  // ---------------- encoder ----------------
  launch_conv(stream, act0, Wp(eInW), WB(eInW), Wp(eInB), nullptr, act1, nullptr,
              64, 512, 1024, 1024, 3, 1, 1, 1, F_POSTRELU);
  cur = act1; t1 = act2; t2 = act0;
  int L = 1024;
  for (int i = 0; i < 3; ++i) {
    launch_conv(stream, cur, Wp(eDownW[i]), WB(eDownW[i]), Wp(eDownB[i]), nullptr, t1, nullptr,
                512, 512, L, L / 2, 4, 2, 1, 1, 0);
    step(); L /= 2;
    for (int j = 0; j < 3; ++j) {
      int d = ENC_DIL[j];
      launch_conv(stream, cur, Wp(eC1W[i][j]), WB(eC1W[i][j]), Wp(eC1B[i][j]), nullptr, t1, nullptr,
                  512, 512, L, L, 3, 1, d, d, F_PRERELU | F_POSTRELU);
      launch_conv(stream, t1, Wp(eC2W[i][j]), WB(eC2W[i][j]), Wp(eC2B[i][j]), cur, t2, nullptr,
                  512, 512, L, L, 1, 1, 0, 1, 0);
      unsigned short* oc = cur; cur = t2; t2 = t1; t1 = oc;
    }
  }
  launch_conv(stream, cur, Wp(eOutW), WB(eOutW), Wp(eOutB), nullptr, nullptr, xenc,
              512, 512, 128, 128, 3, 1, 1, 1, 0);                    // x_enc fp32

  // ---------------- quantize + rotation trick ----------------
  fsq_rot<<<4 * 512, 128, 0, stream>>>(xenc, (const float*)d_in[iLev], act0, used);

  // ---------------- decoder ----------------
  launch_conv(stream, act0, Wp(dInW), WB(dInW), Wp(dInB), nullptr, act1, nullptr,
              512, 512, 128, 128, 3, 1, 1, 1, F_POSTRELU);
  cur = act1; t1 = act2; t2 = act0;
  L = 128;
  for (int i = 0; i < 3; ++i) {
    for (int j = 0; j < 3; ++j) {
      int d = DEC_DIL[j];
      launch_conv(stream, cur, Wp(dC1W[i][j]), WB(dC1W[i][j]), Wp(dC1B[i][j]), nullptr, t1, nullptr,
                  512, 512, L, L, 3, 1, d, d, F_PRERELU | F_POSTRELU);
      launch_conv(stream, t1, Wp(dC2W[i][j]), WB(dC2W[i][j]), Wp(dC2B[i][j]), cur, t2, nullptr,
                  512, 512, L, L, 1, 1, 0, 1, 0);
      unsigned short* oc = cur; cur = t2; t2 = t1; t1 = oc;
    }
    launch_conv(stream, cur, Wp(dUpW[i]), WB(dUpW[i]), Wp(dUpB[i]), nullptr, t1, nullptr,
                512, 512, L, 2 * L, 3, 1, 1, 1, F_UP2);              // repeat(2) fused
    step(); L *= 2;
  }
  launch_conv(stream, cur, Wp(dMidW), WB(dMidW), Wp(dMidB), nullptr, t1, nullptr,
              512, 512, 1024, 1024, 3, 1, 1, 1, F_POSTRELU);
  launch_conv(stream, t1, Wp(dOutW), WB(dOutW), Wp(dOutB), nullptr, nullptr, (float*)d_out,
              512, 64, 1024, 1024, 3, 1, 1, 1, F_TROUT);            // NCH->NHC fused

  finalize_k<<<1, 32, 0, stream>>>((float*)d_out, used, out_size - 2);
}
